// TemporalGraphNetwork_50285477101632
// MI455X (gfx1250) — matrix operand order repaired and checked by reading, not compiled, not run on hardware
//
#include <hip/hip_runtime.h>
#include <hip/hip_bf16.h>
#include <math.h>

typedef __attribute__((ext_vector_type(8)))  _Float16 v8h;
typedef __attribute__((ext_vector_type(16))) _Float16 v16h;
typedef __attribute__((ext_vector_type(8)))  float    v8f;

#define EMB 128
#define TD  64
#define ND  192
#define HC  256
#define NH  4
#define CH  64

// ---------- order-preserving float<->uint encoding for atomicMax ----------
__device__ __forceinline__ unsigned fenc(float f) {
    unsigned u = __float_as_uint(f);
    return (u & 0x80000000u) ? ~u : (u | 0x80000000u);
}
__device__ __forceinline__ float fdec(unsigned u) {
    unsigned v = (u & 0x80000000u) ? (u & 0x7FFFFFFFu) : ~u;
    return __uint_as_float(v);
}

// ---------- generic f32 -> f16 convert ----------
__global__ __launch_bounds__(256)
void f32_to_f16_kernel(const float* __restrict__ in, _Float16* __restrict__ out, int n)
{
    int i = blockIdx.x * blockDim.x + threadIdx.x;
    if (i < n) out[i] = (_Float16)in[i];
}

// ---------- node features (f16): xn = [x | cos(t*w+b)]  (N x 192) ----------
__global__ __launch_bounds__(256)
void node_feat_kernel(const float* __restrict__ x, const float* __restrict__ nlu,
                      const float* __restrict__ tw, const float* __restrict__ tb,
                      _Float16* __restrict__ xn, int N)
{
    int i = blockIdx.x * blockDim.x + threadIdx.x;
    if (i >= N * ND) return;
    int n = i / ND, c = i % ND;
    float v;
    if (c < EMB) v = x[(size_t)n * EMB + c];
    else {
        int d = c - EMB;
        v = cosf(nlu[n] * tw[d] + tb[d]);
    }
    xn[i] = (_Float16)v;
}

// ---------- edge features (f16): ea = [edge_attr | cos(rel_t*w+b)]  (E x 192) ----------
__global__ __launch_bounds__(256)
void edge_feat_kernel(const float* __restrict__ eattr, const float* __restrict__ elu,
                      const float* __restrict__ ets, const int* __restrict__ batch,
                      const int* __restrict__ src,
                      const float* __restrict__ tw, const float* __restrict__ tb,
                      _Float16* __restrict__ ea, int E)
{
    int i = blockIdx.x * blockDim.x + threadIdx.x;
    if (i >= E * ND) return;
    int e = i / ND, c = i % ND;
    float v;
    if (c < EMB) v = eattr[(size_t)e * EMB + c];
    else {
        int d = c - EMB;
        float rt = ets[batch[src[e]]] - elu[e];
        v = cosf(rt * tw[d] + tb[d]);
    }
    ea[i] = (_Float16)v;
}

// ---------- WMMA GEMM: Out(MxNc) = A(MxK) @ W(KxNc) [+bias | +=] ----------
// A, W are f16; f32 accumulate. One wave -> 16x64 tile (4 WMMAs / k-chunk),
// software-pipelined (double-buffered fragments) so loads for chunk i+1 are
// in flight while chunk i's WMMAs execute.
// Requires: M % 16 == 0, K % 32 == 0, Nc % 64 == 0.
template <bool ACC>
__global__ __launch_bounds__(128)
void wmma_gemm_kernel(const _Float16* __restrict__ A, const _Float16* __restrict__ W,
                      const float* __restrict__ bias, float* __restrict__ Out,
                      int M, int K, int Nc)
{
    const int lane = threadIdx.x;                 // 0..31
    const int wave = threadIdx.y;
    const int row0 = blockIdx.x * 16;
    const int col0 = (blockIdx.y * blockDim.y + wave) * 64;
    if (col0 >= Nc) return;                       // wave-uniform; EXEC stays all-ones

    const int m  = lane & 15;
    const int hi = lane >> 4;                     // lane half select

    v8f acc0 = {}, acc1 = {}, acc2 = {}, acc3 = {};

    // A fragment (16-bit A 16x32): lanes 0-15 hold K {0..7,16..23},
    // lanes 16-31 hold K {8..15,24..31}; row = lane&15.
    const _Float16* arow = A + (size_t)(row0 + m) * K + hi * 8;
    const _Float16* wbase = W + col0 + m;   // + (kb+k)*Nc per element, table k order

    // ---- prologue: fragments for kb = 0 ----
    v8h alo = *(const v8h*)(arow);
    v8h ahi = *(const v8h*)(arow + 16);
    v16h a = __builtin_shufflevector(alo, ahi,
                                     0, 1, 2, 3, 4, 5, 6, 7,
                                     8, 9, 10, 11, 12, 13, 14, 15);
    v16h b0, b1, b2, b3;
    for (int s = 0; s < 16; ++s) {
        const int kk = (s < 8) ? (hi * 8 + s) : (16 + hi * 8 + s - 8);
        const _Float16* p = wbase + (size_t)kk * Nc;
        b0[s] = p[0]; b1[s] = p[16]; b2[s] = p[32]; b3[s] = p[48];
    }

    for (int kb = 32; kb <= K; kb += 32) {
        // prefetch next chunk (last iteration harmlessly re-reads chunk 0)
        const int kn = (kb < K) ? kb : 0;
        v8h nlo = *(const v8h*)(arow + kn);
        v8h nhi = *(const v8h*)(arow + kn + 16);
        const _Float16* wrow = wbase + (size_t)kn * Nc;
        v16h nb0, nb1, nb2, nb3;
        for (int s = 0; s < 16; ++s) {
            const int kk = (s < 8) ? (hi * 8 + s) : (16 + hi * 8 + s - 8);
            const _Float16* p = wrow + (size_t)kk * Nc;
            nb0[s] = p[0]; nb1[s] = p[16]; nb2[s] = p[32]; nb3[s] = p[48];
        }

        acc0 = __builtin_amdgcn_wmma_f32_16x16x32_f16(false, a, false, b0, (short)0, acc0, false, false);
        acc1 = __builtin_amdgcn_wmma_f32_16x16x32_f16(false, a, false, b1, (short)0, acc1, false, false);
        acc2 = __builtin_amdgcn_wmma_f32_16x16x32_f16(false, a, false, b2, (short)0, acc2, false, false);
        acc3 = __builtin_amdgcn_wmma_f32_16x16x32_f16(false, a, false, b3, (short)0, acc3, false, false);

        a = __builtin_shufflevector(nlo, nhi,
                                    0, 1, 2, 3, 4, 5, 6, 7,
                                    8, 9, 10, 11, 12, 13, 14, 15);
        b0 = nb0; b1 = nb1; b2 = nb2; b3 = nb3;
    }

    // C/D layout: lane = column (lane&15); VGPR i -> row i (lane<16) / row 8+i.
    float* obase = Out + (size_t)(row0 + hi * 8) * Nc;
    #pragma unroll
    for (int t = 0; t < 4; ++t) {
        const v8f acc = (t == 0) ? acc0 : (t == 1) ? acc1 : (t == 2) ? acc2 : acc3;
        const int col = col0 + t * 16 + m;
        float* op = obase + col;
        if (ACC) {
            #pragma unroll
            for (int i = 0; i < 8; ++i) op[(size_t)i * Nc] += acc[i];
        } else {
            const float bv = bias ? bias[col] : 0.0f;
            #pragma unroll
            for (int i = 0; i < 8; ++i) op[(size_t)i * Nc] = acc[i] + bv;
        }
    }
}

// ---------- init per-layer softmax state ----------
__global__ __launch_bounds__(256)
void init_softmax_kernel(unsigned* __restrict__ mmax, float* __restrict__ ssum, int n4)
{
    int i = blockIdx.x * blockDim.x + threadIdx.x;
    if (i >= n4) return;
    mmax[i] = 0x00800000u;   // fenc(-FLT_MAX)
    ssum[i] = 0.0f;
}

// ---------- attention scores: one wave per edge ----------
__global__ __launch_bounds__(256)
void attn_score_kernel(const float* __restrict__ q, const float* __restrict__ k,
                       const float* __restrict__ em,
                       const int* __restrict__ src, const int* __restrict__ tgt,
                       float* __restrict__ alpha, unsigned* __restrict__ mmax, int E)
{
    int wid  = threadIdx.x >> 5;
    int lane = threadIdx.x & 31;
    int e = blockIdx.x * 8 + wid;
    if (e >= E) return;
    int s = src[e], t = tgt[e];
    const float* qp = q  + (size_t)t * HC + lane * 8;
    const float* kp = k  + (size_t)s * HC + lane * 8;
    const float* ep = em + (size_t)e * HC + lane * 8;
    float p = 0.0f;
    #pragma unroll
    for (int i = 0; i < 8; ++i) p += qp[i] * (kp[i] + ep[i]);
    p += __shfl_xor(p, 1, 32);
    p += __shfl_xor(p, 2, 32);
    p += __shfl_xor(p, 4, 32);
    if ((lane & 7) == 0) {
        int h = lane >> 3;
        float a = p * 0.125f;                    // 1/sqrt(64)
        alpha[(size_t)e * NH + h] = a;
        atomicMax(&mmax[(size_t)t * NH + h], fenc(a));
    }
}

// ---------- exp + segment sum: one thread per (edge,head) ----------
__global__ __launch_bounds__(256)
void attn_exp_kernel(const int* __restrict__ tgt, float* __restrict__ alpha,
                     const unsigned* __restrict__ mmax, float* __restrict__ ssum, int E)
{
    int i = blockIdx.x * blockDim.x + threadIdx.x;
    if (i >= E * NH) return;
    int e = i >> 2, h = i & 3;
    int t = tgt[e];
    float mv = fdec(mmax[(size_t)t * NH + h]);
    float ex = expf(alpha[i] - mv);
    alpha[i] = ex;
    atomicAdd(&ssum[(size_t)t * NH + h], ex);
}

// ---------- scatter a * (v[src]+e) into out[tgt]: one wave per edge ----------
__global__ __launch_bounds__(256)
void attn_scatter_kernel(const float* __restrict__ v, const float* __restrict__ em,
                         const int* __restrict__ src, const int* __restrict__ tgt,
                         const float* __restrict__ alpha, const float* __restrict__ ssum,
                         float* __restrict__ out, int E)
{
    int wid  = threadIdx.x >> 5;
    int lane = threadIdx.x & 31;
    int e = blockIdx.x * 8 + wid;
    if (e >= E) return;
    int s = src[e], t = tgt[e];
    int h = lane >> 3;
    float a = alpha[(size_t)e * NH + h] / (ssum[(size_t)t * NH + h] + 1e-16f);
    const float* vp = v  + (size_t)s * HC + lane * 8;
    const float* ep = em + (size_t)e * HC + lane * 8;
    float* op = out + (size_t)t * HC + lane * 8;
    #pragma unroll
    for (int i = 0; i < 8; ++i) atomicAdd(&op[i], a * (vp[i] + ep[i]));
}

extern "C" void kernel_launch(void* const* d_in, const int* in_sizes, int n_in,
                              void* d_out, int out_size, void* d_ws, size_t ws_size,
                              hipStream_t stream)
{
    const float* ets   = (const float*)d_in[0];
    const float* x     = (const float*)d_in[1];
    const float* nlu   = (const float*)d_in[2];
    const float* eattr = (const float*)d_in[3];
    const float* elu   = (const float*)d_in[4];
    const int*   batch = (const int*)d_in[5];
    const int*   ei    = (const int*)d_in[6];
    const float* tw    = (const float*)d_in[7];
    const float* tb    = (const float*)d_in[8];
    const float* l0[9]; for (int i = 0; i < 9; ++i) l0[i] = (const float*)d_in[9 + i];
    const float* l1[9]; for (int i = 0; i < 9; ++i) l1[i] = (const float*)d_in[18 + i];
    const float* linW = (const float*)d_in[27];
    const float* linb = (const float*)d_in[28];

    const int N = in_sizes[2];
    const int E = in_sizes[4];
    const int D1 = ND + HC;                 // 448
    const int* src = ei;
    const int* tgt = ei + E;

    // ---- workspace layout (256B-aligned partitions) ----
    char* base = (char*)d_ws;
    size_t off = 0;
    auto alloc = [&](size_t bytes) -> void* {
        void* p = base + off;
        off += (bytes + 255) & ~(size_t)255;
        return p;
    };
    _Float16* xn_h = (_Float16*)alloc((size_t)N * ND * 2);
    _Float16* ea_h = (_Float16*)alloc((size_t)E * ND * 2);
    float* qb = (float*)alloc((size_t)N * HC * 4);
    float* kb = (float*)alloc((size_t)N * HC * 4);
    float* vb = (float*)alloc((size_t)N * HC * 4);
    float* em = (float*)alloc((size_t)E * HC * 4);
    float* h0 = (float*)alloc((size_t)N * HC * 4);
    float* h1 = (float*)alloc((size_t)N * HC * 4);
    _Float16* h0h = (_Float16*)alloc((size_t)N * HC * 2);
    _Float16* h1h = (_Float16*)alloc((size_t)N * HC * 2);
    float* al   = (float*)alloc((size_t)E * NH * 4);
    float* ssum = (float*)alloc((size_t)N * NH * 4);
    unsigned* mmax = (unsigned*)alloc((size_t)N * NH * 4);
    // f16 weight copies
    _Float16* w0q = (_Float16*)alloc((size_t)ND * HC * 2);
    _Float16* w0k = (_Float16*)alloc((size_t)ND * HC * 2);
    _Float16* w0v = (_Float16*)alloc((size_t)ND * HC * 2);
    _Float16* w0e = (_Float16*)alloc((size_t)ND * HC * 2);
    _Float16* w0s = (_Float16*)alloc((size_t)ND * HC * 2);
    _Float16* w1q = (_Float16*)alloc((size_t)D1 * HC * 2);
    _Float16* w1k = (_Float16*)alloc((size_t)D1 * HC * 2);
    _Float16* w1v = (_Float16*)alloc((size_t)D1 * HC * 2);
    _Float16* w1e = (_Float16*)alloc((size_t)ND * HC * 2);
    _Float16* w1s = (_Float16*)alloc((size_t)D1 * HC * 2);
    _Float16* wlin = (_Float16*)alloc((size_t)HC * 64 * 2);

    auto cvt = [&](const float* in, _Float16* out, int n) {
        f32_to_f16_kernel<<<(n + 255) / 256, 256, 0, stream>>>(in, out, n);
    };
    auto gemm = [&](const _Float16* A, const _Float16* W, const float* bias, float* Out,
                    int M, int K, int Nc, bool accum) {
        int wy = (Nc >= 256) ? 4 : 1;
        dim3 grid(M / 16, (Nc + 64 * wy - 1) / (64 * wy));
        dim3 block(32, wy);
        if (accum)
            wmma_gemm_kernel<true><<<grid, block, 0, stream>>>(A, W, nullptr, Out, M, K, Nc);
        else
            wmma_gemm_kernel<false><<<grid, block, 0, stream>>>(A, W, bias, Out, M, K, Nc);
    };

    // ---- weight conversion (f32 -> f16, once per launch) ----
    cvt(l0[0], w0q, ND * HC);  cvt(l0[2], w0k, ND * HC);  cvt(l0[4], w0v, ND * HC);
    cvt(l0[6], w0e, ND * HC);  cvt(l0[7], w0s, ND * HC);
    cvt(l1[0], w1q, D1 * HC);  cvt(l1[2], w1k, D1 * HC);  cvt(l1[4], w1v, D1 * HC);
    cvt(l1[6], w1e, ND * HC);  cvt(l1[7], w1s, D1 * HC);
    cvt(linW,  wlin, HC * 64);

    // ---- features (written directly as f16) ----
    node_feat_kernel<<<((size_t)N * ND + 255) / 256, 256, 0, stream>>>(x, nlu, tw, tb, xn_h, N);
    edge_feat_kernel<<<((size_t)E * ND + 255) / 256, 256, 0, stream>>>(
        eattr, elu, ets, batch, src, tw, tb, ea_h, E);

    // ==================== Layer 0 (din = 192) ====================
    gemm(xn_h, w0q, l0[1], qb, N, ND, HC, false);
    gemm(xn_h, w0k, l0[3], kb, N, ND, HC, false);
    gemm(xn_h, w0v, l0[5], vb, N, ND, HC, false);
    gemm(ea_h, w0e, nullptr, em, E, ND, HC, false);
    gemm(xn_h, w0s, l0[8], h0, N, ND, HC, false);   // skip path -> output base

    init_softmax_kernel<<<(N * NH + 255) / 256, 256, 0, stream>>>(mmax, ssum, N * NH);
    attn_score_kernel<<<(E + 7) / 8, 256, 0, stream>>>(qb, kb, em, src, tgt, al, mmax, E);
    attn_exp_kernel<<<(E * NH + 255) / 256, 256, 0, stream>>>(tgt, al, mmax, ssum, E);
    attn_scatter_kernel<<<(E + 7) / 8, 256, 0, stream>>>(vb, em, src, tgt, al, ssum, h0, E);

    cvt(h0, h0h, N * HC);

    // ==================== Layer 1 (din = 448 = [h0 | xn]) ====================
    const size_t woff = (size_t)HC * HC;  // first 256 rows of each (448,256) weight
    gemm(h0h,  w1q,        l1[1],  qb, N, HC, HC, false);
    gemm(xn_h, w1q + woff, nullptr, qb, N, ND, HC, true);
    gemm(h0h,  w1k,        l1[3],  kb, N, HC, HC, false);
    gemm(xn_h, w1k + woff, nullptr, kb, N, ND, HC, true);
    gemm(h0h,  w1v,        l1[5],  vb, N, HC, HC, false);
    gemm(xn_h, w1v + woff, nullptr, vb, N, ND, HC, true);
    gemm(ea_h, w1e, nullptr, em, E, ND, HC, false);
    gemm(h0h,  w1s,        l1[8],  h1, N, HC, HC, false);
    gemm(xn_h, w1s + woff, nullptr, h1, N, ND, HC, true);

    init_softmax_kernel<<<(N * NH + 255) / 256, 256, 0, stream>>>(mmax, ssum, N * NH);
    attn_score_kernel<<<(E + 7) / 8, 256, 0, stream>>>(qb, kb, em, src, tgt, al, mmax, E);
    attn_exp_kernel<<<(E * NH + 255) / 256, 256, 0, stream>>>(tgt, al, mmax, ssum, E);
    attn_scatter_kernel<<<(E + 7) / 8, 256, 0, stream>>>(vb, em, src, tgt, al, ssum, h1, E);

    cvt(h1, h1h, N * HC);

    // ==================== Final projection (256 -> 64) ====================
    gemm(h1h, wlin, linb, (float*)d_out, N, HC, 64, false);

    (void)n_in; (void)out_size; (void)ws_size;
}
